// BitConv2d_55791625175521
// MI455X (gfx1250) — matrix operand order repaired and checked by reading, not compiled, not run on hardware
//
#include <hip/hip_runtime.h>

typedef __attribute__((ext_vector_type(16))) _Float16 v16h;
typedef __attribute__((ext_vector_type(8)))  _Float16 v8h;
typedef __attribute__((ext_vector_type(8)))  float    v8f;
typedef __attribute__((ext_vector_type(4)))  int      v4i;

#define N_IMG  32
#define C_IN   256
#define H_IMG  56
#define W_IMG  56
#define C_OUT  256
#define HW     (H_IMG * W_IMG)      // 3136
#define NPIX   (N_IMG * HW)         // 100352
#define KDIM   (C_IN * 9)           // 2304
#define TILE_M 128
#define TILE_N 128

// padded activation planes (1-px zero border -> no bounds checks in gather)
#define PW   58
#define PH   58
#define PHW  (PW * PH)              // 3364

// fast path tiling
#define KSTEP64 64
#define LDSPA   72                  // padded row (halves); 144B stride = 9*16B

// fallback tiling
#define KSTEP 32
#define LDSP  40

// workspace layout
#define WQ_OFF   256
#define WQ_BYTES (C_OUT * KDIM * 2)                   // 1,179,648
#define XQ_OFF   (WQ_OFF + WQ_BYTES)                  // 1,179,904 (256-aligned)
#define XQ_ELEMS (N_IMG * C_IN * PHW)                 // 27,557,888
#define XQ_BYTES ((size_t)XQ_ELEMS * 2)
#define WS_NEED  ((size_t)XQ_OFF + XQ_BYTES)

#define AS1 __attribute__((address_space(1)))
#define AS3 __attribute__((address_space(3)))

#if defined(__has_builtin)
#  if __has_builtin(__builtin_amdgcn_global_load_async_to_lds_b128) && \
      __has_builtin(__builtin_amdgcn_s_wait_asynccnt)
#    define HAVE_ASYNC_LDS 1
#  endif
#endif
#ifndef HAVE_ASYNC_LDS
#  define HAVE_ASYNC_LDS 0
#endif

// ---------------- abs-max reduction ----------------
__global__ void absmax_init(unsigned* p) { *p = 0u; }

__global__ __launch_bounds__(256)
void absmax_kernel(const float* __restrict__ x, unsigned* __restrict__ out, int n4) {
  __shared__ float red[256];
  int tid = blockIdx.x * blockDim.x + threadIdx.x;
  int stride = gridDim.x * blockDim.x;
  const float4* x4 = (const float4*)x;
  float m = 0.0f;
  for (int i = tid; i < n4; i += stride) {
    float4 v = x4[i];
    m = fmaxf(m, fmaxf(fmaxf(fabsf(v.x), fabsf(v.y)), fmaxf(fabsf(v.z), fabsf(v.w))));
  }
  red[threadIdx.x] = m;
  __syncthreads();
  for (int s = 128; s > 0; s >>= 1) {
    if (threadIdx.x < s) red[threadIdx.x] = fmaxf(red[threadIdx.x], red[threadIdx.x + s]);
    __syncthreads();
  }
  if (threadIdx.x == 0) atomicMax(out, __float_as_uint(red[0]));  // all >=0: uint order == float order
}

// ---------------- chimera weight quantization (forward value) ----------------
__device__ __forceinline__ _Float16 quant_w(float w, float inv_gamma) {
  float ws = w * inv_gamma;
  float t  = rintf(ws);
  t  = fminf(fmaxf(t, -1.0f), 1.0f);
  float wq = 0.3f * ws + 0.7f * t;          // (1-ALPHA)*w_scaled + ALPHA*ternary
  wq = fminf(fmaxf(wq, -1.0f), 1.0f);       // STE clip (forward value)
  return (_Float16)wq;
}

// ---------------- precompute: w_q -> fp16, row-major [Cout][KDIM] ----------------
__global__ __launch_bounds__(256)
void quant_w_kernel(const float* __restrict__ w, _Float16* __restrict__ wq,
                    const float* __restrict__ scale_ema) {
  const float inv_gamma = 1.0f / fmaxf(*scale_ema, 1e-6f);
  int i = blockIdx.x * 256 + threadIdx.x;   // exact grid: C_OUT*KDIM/256
  wq[i] = quant_w(w[i], inv_gamma);
}

// ---------------- precompute: x_q -> fp16, padded 58x58 planes ----------------
__global__ __launch_bounds__(256)
void quant_x_kernel(const float* __restrict__ x, _Float16* __restrict__ xq,
                    const unsigned* __restrict__ absbits) {
  const float beta = __uint_as_float(*absbits) * (1.0f / 127.0f) + 1e-6f;
  const float inv_beta = 1.0f / beta;
  unsigned i = blockIdx.x * 256 + threadIdx.x;   // exact grid: XQ_ELEMS/256
  unsigned plane = i / PHW;
  unsigned r = i - plane * PHW;
  unsigned h = r / PW;
  unsigned w = r - h * PW;
  float v = 0.0f;
  if (h >= 1u && h <= (unsigned)H_IMG && w >= 1u && w <= (unsigned)W_IMG)
    v = x[(size_t)plane * HW + (h - 1u) * W_IMG + (w - 1u)];
  float q = fminf(fmaxf(rintf(v * inv_beta), -127.0f), 127.0f);
  xq[i] = (_Float16)q;
}

// ---------------- FAST implicit-GEMM conv: prequantized fp16 operands ----------------
__global__ __launch_bounds__(256)
void bitconv_wmma_fast(const _Float16* __restrict__ xq, const _Float16* __restrict__ wq,
                       const float* __restrict__ bias, const float* __restrict__ scale_ema,
                       const unsigned* __restrict__ absbits, float* __restrict__ y) {
  __shared__ _Float16 As[TILE_M][LDSPA];   // weight tile [co_local][k_local]
  __shared__ _Float16 Bs[TILE_N][LDSPA];   // pixel tile  [p_local][k_local]
  __shared__ int pixbase[TILE_N];

  const int tid   = threadIdx.x;
  const int lane  = tid & 31;
  const int wv    = tid >> 5;
  const int wm    = wv & 1;
  const int wn    = wv >> 1;
  const int lhalf = lane >> 4;
  const int l16   = lane & 15;

  const int pix_base = blockIdx.x * TILE_N;
  const int co_base  = blockIdx.y * TILE_M;

  // per-pixel base offset into padded xq (computed once)
  if (tid < TILE_N) {
    int p  = pix_base + tid;
    int n  = p / HW;
    int hw = p - n * HW;
    int h  = hw / W_IMG;
    int w  = hw - h * W_IMG;
    pixbase[tid] = n * (C_IN * PHW) + h * PW + w;
  }

  const float beta     = __uint_as_float(*absbits) * (1.0f / 127.0f) + 1e-6f;
  const float gamma    = fmaxf(*scale_ema, 1e-6f);
  const float outscale = beta * gamma;

  v8f acc[4][2];
  const v8f vzero = {0.f, 0.f, 0.f, 0.f, 0.f, 0.f, 0.f, 0.f};
#pragma unroll
  for (int mi = 0; mi < 4; ++mi)
#pragma unroll
    for (int ni = 0; ni < 2; ++ni) acc[mi][ni] = vzero;

  // A loader: 2 threads per row, 32 consecutive halves (64B) each
  const int a_row = tid >> 1;
  const int a_col = (tid & 1) * 32;
  const _Float16* wrow = wq + (size_t)(co_base + a_row) * KDIM + a_col;
  _Float16* a_dst = &As[a_row][a_col];

  // B loader: one k per thread, 32 consecutive pixels
  const int bk  = tid & 63;
  const int bp0 = (tid >> 6) * 32;

  __syncthreads();   // pixbase ready

  for (int kk = 0; kk < KDIM; kk += KSTEP64) {
    // ---- A tile: 64B per thread, async global->LDS if available ----
#if HAVE_ASYNC_LDS
    {
      AS1 v4i* gsrc = (AS1 v4i*)(wrow + kk);
      AS3 v4i* ldst = (AS3 v4i*)a_dst;
      __builtin_amdgcn_global_load_async_to_lds_b128(gsrc,     ldst,     0, 0);
      __builtin_amdgcn_global_load_async_to_lds_b128(gsrc + 1, ldst + 1, 0, 0);
      __builtin_amdgcn_global_load_async_to_lds_b128(gsrc + 2, ldst + 2, 0, 0);
      __builtin_amdgcn_global_load_async_to_lds_b128(gsrc + 3, ldst + 3, 0, 0);
    }
#else
    {
      const uint4* s = (const uint4*)(wrow + kk);
      uint4* d = (uint4*)a_dst;
      d[0] = s[0]; d[1] = s[1]; d[2] = s[2]; d[3] = s[3];
    }
#endif
    // ---- B tile: padded gather, no bounds checks, no quantize ----
    {
      const int k  = kk + bk;
      const int ci = k / 9;
      const int r  = k - ci * 9;
      const int kh = r / 3;
      const int kw = r - kh * 3;
      const int koff = ci * PHW + kh * PW + kw;
#pragma unroll
      for (int pi = 0; pi < 32; ++pi) {
        Bs[bp0 + pi][bk] = xq[pixbase[bp0 + pi] + koff];
      }
    }
#if HAVE_ASYNC_LDS
    __builtin_amdgcn_s_wait_asynccnt(0);
#endif
    __syncthreads();

    // ---- two K=32 WMMA sub-steps ----
#pragma unroll
    for (int ks = 0; ks < KSTEP64; ks += 32) {
      v16h afrag[4];
#pragma unroll
      for (int mi = 0; mi < 4; ++mi) {
        const int m = wm * 64 + mi * 16 + l16;                   // A: M = lane&15
        v8h lo = *(const v8h*)&As[m][ks + lhalf * 16];           // K = 16*lhalf + 0..7
        v8h hi = *(const v8h*)&As[m][ks + lhalf * 16 + 8];       // K = 16*lhalf + 8..15
        afrag[mi] = __builtin_shufflevector(lo, hi, 0, 1, 2, 3, 4, 5, 6, 7,
                                            8, 9, 10, 11, 12, 13, 14, 15);
      }
      v16h bfrag[2];
#pragma unroll
      for (int ni = 0; ni < 2; ++ni) {
        const int pr = wn * 32 + ni * 16 + l16;                  // B: N = lane&15
        v8h lo = *(const v8h*)&Bs[pr][ks + lhalf * 16];          // K = 16*lhalf + 0..7
        v8h hi = *(const v8h*)&Bs[pr][ks + lhalf * 16 + 8];      // K = 16*lhalf + 8..15
        bfrag[ni] = __builtin_shufflevector(lo, hi, 0, 1, 2, 3, 4, 5, 6, 7,
                                            8, 9, 10, 11, 12, 13, 14, 15);
      }
#pragma unroll
      for (int mi = 0; mi < 4; ++mi)
#pragma unroll
        for (int ni = 0; ni < 2; ++ni)
          acc[mi][ni] = __builtin_amdgcn_wmma_f32_16x16x32_f16(
              false, afrag[mi], false, bfrag[ni], (short)0, acc[mi][ni], false, false);
    }
    __syncthreads();
  }

  // ---- epilogue: y = beta*gamma*acc + bias, scatter to NCHW ----
#pragma unroll
  for (int mi = 0; mi < 4; ++mi) {
#pragma unroll
    for (int j = 0; j < 8; ++j) {
      const int co = co_base + wm * 64 + mi * 16 + lhalf * 8 + j;
      const float bv = bias[co];
#pragma unroll
      for (int ni = 0; ni < 2; ++ni) {
        const int p  = pix_base + wn * 32 + ni * 16 + l16;
        const int n  = p / HW;
        const int hw = p - n * HW;
        y[((size_t)n * C_OUT + co) * HW + hw] = outscale * acc[mi][ni][j] + bv;
      }
    }
  }
}

// ---------------- FALLBACK: fused kernel (tiny workspace) ----------------
__global__ __launch_bounds__(256)
void bitconv_wmma_fused(const float* __restrict__ x, const float* __restrict__ wgt,
                        const float* __restrict__ bias, const float* __restrict__ scale_ema,
                        const unsigned* __restrict__ absmax_bits, float* __restrict__ y) {
  __shared__ _Float16 As[TILE_M][LDSP];
  __shared__ _Float16 Bs[TILE_N][LDSP];

  const int tid   = threadIdx.x;
  const int lane  = tid & 31;
  const int wv    = tid >> 5;
  const int wm    = wv & 1;
  const int wn    = wv >> 1;
  const int lhalf = lane >> 4;
  const int l16   = lane & 15;

  const int pix_base = blockIdx.x * TILE_N;
  const int co_base  = blockIdx.y * TILE_M;

  const float maxabs    = __uint_as_float(*absmax_bits);
  const float beta      = maxabs * (1.0f / 127.0f) + 1e-6f;
  const float inv_beta  = 1.0f / beta;
  const float gamma     = fmaxf(*scale_ema, 1e-6f);
  const float inv_gamma = 1.0f / gamma;
  const float outscale  = beta * gamma;

  v8f acc[4][2];
  const v8f vzero = {0.f, 0.f, 0.f, 0.f, 0.f, 0.f, 0.f, 0.f};
#pragma unroll
  for (int mi = 0; mi < 4; ++mi)
#pragma unroll
    for (int ni = 0; ni < 2; ++ni) acc[mi][ni] = vzero;

  const int a_row = tid >> 1;
  const int a_col = (tid & 1) * 16;
  const float* wrow = wgt + (size_t)(co_base + a_row) * KDIM;

  const int bk  = tid & 31;
  const int bp0 = (tid >> 5) * 16;

  for (int kk = 0; kk < KDIM; kk += KSTEP) {
    {
      const float4* wp = (const float4*)(wrow + kk + a_col);
#pragma unroll
      for (int v = 0; v < 4; ++v) {
        float4 w4 = wp[v];
        As[a_row][a_col + v * 4 + 0] = quant_w(w4.x, inv_gamma);
        As[a_row][a_col + v * 4 + 1] = quant_w(w4.y, inv_gamma);
        As[a_row][a_col + v * 4 + 2] = quant_w(w4.z, inv_gamma);
        As[a_row][a_col + v * 4 + 3] = quant_w(w4.w, inv_gamma);
      }
    }
    {
      const int kglob = kk + bk;
      const int ci = kglob / 9;
      const int r  = kglob - ci * 9;
      const int kh = r / 3;
      const int kw = r - kh * 3;
#pragma unroll
      for (int pi = 0; pi < 16; ++pi) {
        const int p  = pix_base + bp0 + pi;
        const int n  = p / HW;
        const int hw = p - n * HW;
        const int h  = hw / W_IMG;
        const int w  = hw - h * W_IMG;
        const int ih = h + kh - 1;
        const int iw = w + kw - 1;
        float xv = 0.0f;
        if ((unsigned)ih < (unsigned)H_IMG && (unsigned)iw < (unsigned)W_IMG)
          xv = x[(((size_t)n * C_IN + ci) * H_IMG + ih) * W_IMG + iw];
        float q = fminf(fmaxf(rintf(xv * inv_beta), -127.0f), 127.0f);
        Bs[bp0 + pi][bk] = (_Float16)q;
      }
    }
    __syncthreads();

    v16h afrag[4];
#pragma unroll
    for (int mi = 0; mi < 4; ++mi) {
      const int m = wm * 64 + mi * 16 + l16;
      v8h lo = *(const v8h*)&As[m][lhalf * 16];
      v8h hi = *(const v8h*)&As[m][lhalf * 16 + 8];
      afrag[mi] = __builtin_shufflevector(lo, hi, 0, 1, 2, 3, 4, 5, 6, 7,
                                          8, 9, 10, 11, 12, 13, 14, 15);
    }
    v16h bfrag[2];
#pragma unroll
    for (int ni = 0; ni < 2; ++ni) {
      const int pr = wn * 32 + ni * 16 + l16;
      v8h lo = *(const v8h*)&Bs[pr][lhalf * 16];
      v8h hi = *(const v8h*)&Bs[pr][lhalf * 16 + 8];
      bfrag[ni] = __builtin_shufflevector(lo, hi, 0, 1, 2, 3, 4, 5, 6, 7,
                                          8, 9, 10, 11, 12, 13, 14, 15);
    }
#pragma unroll
    for (int mi = 0; mi < 4; ++mi)
#pragma unroll
      for (int ni = 0; ni < 2; ++ni)
        acc[mi][ni] = __builtin_amdgcn_wmma_f32_16x16x32_f16(
            false, afrag[mi], false, bfrag[ni], (short)0, acc[mi][ni], false, false);
    __syncthreads();
  }

#pragma unroll
  for (int mi = 0; mi < 4; ++mi) {
#pragma unroll
    for (int j = 0; j < 8; ++j) {
      const int co = co_base + wm * 64 + mi * 16 + lhalf * 8 + j;
      const float bv = bias[co];
#pragma unroll
      for (int ni = 0; ni < 2; ++ni) {
        const int p  = pix_base + wn * 32 + ni * 16 + l16;
        const int n  = p / HW;
        const int hw = p - n * HW;
        y[((size_t)n * C_OUT + co) * HW + hw] = outscale * acc[mi][ni][j] + bv;
      }
    }
  }
}

extern "C" void kernel_launch(void* const* d_in, const int* in_sizes, int n_in,
                              void* d_out, int out_size, void* d_ws, size_t ws_size,
                              hipStream_t stream) {
  const float* x         = (const float*)d_in[0];
  const float* weight    = (const float*)d_in[1];
  const float* bias      = (const float*)d_in[2];
  const float* scale_ema = (const float*)d_in[3];
  float* y = (float*)d_out;
  unsigned* absbits = (unsigned*)d_ws;

  absmax_init<<<1, 1, 0, stream>>>(absbits);
  absmax_kernel<<<2048, 256, 0, stream>>>(x, absbits, (N_IMG * C_IN * HW) / 4);

  dim3 grid(NPIX / TILE_N, C_OUT / TILE_M);  // 784 x 2

  if (ws_size >= WS_NEED) {
    _Float16* wqp = (_Float16*)((char*)d_ws + WQ_OFF);
    _Float16* xqp = (_Float16*)((char*)d_ws + XQ_OFF);
    quant_w_kernel<<<(C_OUT * KDIM) / 256, 256, 0, stream>>>(weight, wqp, scale_ema);
    quant_x_kernel<<<XQ_ELEMS / 256, 256, 0, stream>>>(x, xqp, absbits);
    bitconv_wmma_fast<<<grid, 256, 0, stream>>>(xqp, wqp, bias, scale_ema, absbits, y);
  } else {
    bitconv_wmma_fused<<<grid, 256, 0, stream>>>(x, weight, bias, scale_ema, absbits, y);
  }
}
